// DpSe_9380208575068
// MI455X (gfx1250) — matrix operand order repaired and checked by reading, not compiled, not run on hardware
//
#include <hip/hip_runtime.h>
#include <hip/hip_bf16.h>
#include <cstdint>

typedef __attribute__((ext_vector_type(16))) _Float16 v16h;
typedef __attribute__((ext_vector_type(8)))  float    v8f;

#define BATCH 2
#define NPA   1024           // atoms per batch
#define NA    2048           // BATCH*NPA
#define MN    128            // neighbors per atom (both type blocks)
#define MBLK  64             // neighbors per type block
#define NTY   2
#define FITW  240
#define NP    256            // padded N (output cols) for 240-wide layers
#define KP    256            // padded K for 240-wide operands
#define DDIM  1600
#define RMINc 3.0f
#define RMAXc 6.0f

// ---------------------------------------------------------------- utilities
__global__ void dpse_zero(float* p, int n) {
  for (int i = blockIdx.x * blockDim.x + threadIdx.x; i < n; i += gridDim.x * blockDim.x) p[i] = 0.f;
}

// dst[n*Kp+k] = (k<K && n<N) ? src[k*N+n] : 0   (f32 K x N -> f16 Np x Kp, transposed+padded)
__global__ void dpse_cvtT(const float* __restrict__ src, _Float16* __restrict__ dst,
                          int K, int N, int Kp, int total) {
  int i = blockIdx.x * blockDim.x + threadIdx.x;
  if (i >= total) return;
  int n = i / Kp, k = i % Kp;
  dst[i] = (k < K && n < N) ? (_Float16)src[k * N + n] : (_Float16)0.f;
}

// dst[r*Cp+c] = (r<R && c<C) ? src[r*C+c] : 0   (plain copy, row+col padded)
__global__ void dpse_cvtP(const float* __restrict__ src, _Float16* __restrict__ dst,
                          int R, int C, int Cp, int total) {
  int i = blockIdx.x * blockDim.x + threadIdx.x;
  if (i >= total) return;
  int r = i / Cp, c = i % Cp;
  dst[i] = (r < R && c < C) ? (_Float16)src[r * C + c] : (_Float16)0.f;
}

// ---------------------------------------------------------------- calc_Ri
__global__ void dpse_calc_ri(const float* __restrict__ idr,
                             float* __restrict__ Ri, float* __restrict__ Rid) {
  int row = blockIdx.x * blockDim.x + threadIdx.x;       // NA*MN rows
  if (row >= NA * MN) return;
  float r  = idr[row * 4 + 0];
  float d0 = idr[row * 4 + 1], d1 = idr[row * 4 + 2], d2 = idr[row * 4 + 3];
  float inv = 1.f / r;
  float u = (r - RMINc) / (RMAXc - RMINc);
  float poly  = u * u * u * (-6.f * u * u + 15.f * u - 10.f) + 1.f;
  float dpoly = -30.f * u * u * (u - 1.f) * (u - 1.f) / (RMAXc - RMINc);
  float s, sp;
  if (r < RMINc)      { s = inv;        sp = -inv * inv; }
  else if (r < RMAXc) { s = inv * poly; sp = -inv * inv * poly + inv * dpoly; }
  else                { s = 0.f;        sp = 0.f; }
  float rh[3] = { d0 * inv, d1 * inv, d2 * inv };
  Ri[row * 4 + 0] = s;
  Ri[row * 4 + 1] = s * rh[0];
  Ri[row * 4 + 2] = s * rh[1];
  Ri[row * 4 + 3] = s * rh[2];
  float coef = sp - s * inv, sir = s * inv;
  float* rd = Rid + row * 12;
  #pragma unroll
  for (int c = 0; c < 3; ++c) rd[c] = sp * rh[c];
  #pragma unroll
  for (int f = 0; f < 3; ++f)
    #pragma unroll
    for (int c = 0; c < 3; ++c)
      rd[(f + 1) * 3 + c] = coef * rh[f] * rh[c] + ((f == c) ? sir : 0.f);
}

// ------------------------------------------------- embedding fwd + scat + D
__global__ void __launch_bounds__(64)
dpse_embed_fwd(const float* __restrict__ Ri,
               const float* __restrict__ EW0, const float* __restrict__ Eb0,
               const float* __restrict__ EW1, const float* __restrict__ Eb1,
               const float* __restrict__ EW2, const float* __restrict__ Eb2,
               float* __restrict__ scatg, _Float16* __restrict__ Dh) {
  __shared__ float sW0[25], sB0[25], sW1[25 * 50], sB1[50], sW2[50 * 100], sB2[100];
  __shared__ float sscat[400];
  __shared__ float sx1[64 * 25], sx2[64 * 50];
  int t    = blockIdx.x & 1;
  int atom = blockIdx.x >> 1;
  int m    = threadIdx.x;
  for (int i = m; i < 400; i += 64) sscat[i] = 0.f;
  for (int j = 0; j < NTY; ++j) {
    __syncthreads();
    int p = t * NTY + j;
    for (int i = m; i < 1250; i += 64) sW1[i] = EW1[p * 1250 + i];
    for (int i = m; i < 5000; i += 64) sW2[i] = EW2[p * 5000 + i];
    if (m < 25) { sW0[m] = EW0[p * 25 + m]; sB0[m] = Eb0[p * 25 + m]; }
    if (m < 50) sB1[m] = Eb1[p * 50 + m];
    for (int i = m; i < 100; i += 64) sB2[i] = Eb2[p * 100 + i];
    __syncthreads();
    int nrow = atom * MN + j * MBLK + m;
    float blk[4];
    #pragma unroll
    for (int f = 0; f < 4; ++f) blk[f] = Ri[nrow * 4 + f];
    float s = blk[0];
    for (int k = 0; k < 25; ++k) sx1[m * 25 + k] = tanhf(s * sW0[k] + sB0[k]);
    for (int g = 0; g < 50; ++g) {
      float a = sB1[g];
      for (int k = 0; k < 25; ++k) a += sx1[m * 25 + k] * sW1[k * 50 + g];
      sx2[m * 50 + g] = tanhf(a) + sx1[m * 25 + (g % 25)];
    }
    for (int g = 0; g < 100; ++g) {
      float a = sB2[g];
      for (int k = 0; k < 50; ++k) a += sx2[m * 50 + k] * sW2[k * 100 + g];
      float G = tanhf(a) + sx2[m * 50 + (g % 50)];
      #pragma unroll
      for (int f = 0; f < 4; ++f) atomicAdd(&sscat[f * 100 + g], blk[f] * G);
    }
  }
  __syncthreads();
  const float invMN = 1.f / (float)MN;
  for (int i = m; i < 400; i += 64) {
    float v = sscat[i] * invMN;
    sscat[i] = v;
    scatg[((size_t)t * NA + atom) * 400 + i] = v;
  }
  __syncthreads();
  for (int i = m; i < DDIM; i += 64) {
    int g = i >> 4, h = i & 15;
    float d = 0.f;
    #pragma unroll
    for (int f = 0; f < 4; ++f) d += sscat[f * 100 + g] * sscat[f * 100 + h];
    Dh[((size_t)t * NA + atom) * DDIM + i] = (_Float16)d;
  }
}

// ---------------------------------------------------------------- WMMA GEMM
// C(2048 x ldC, f32) = A(2048 x Kp, f16 rowmaj) * BT(Ntiles*16 x Kp, f16 rowmaj)^T
// block = 1 wave; each wave computes a full (guard-free) 16 x 64 output strip.
__global__ void __launch_bounds__(32)
dpse_gemm(const _Float16* __restrict__ A, const _Float16* __restrict__ BT,
          float* __restrict__ C, int ldC, int Kp) {
  int m0 = blockIdx.x * 16;
  int n0 = blockIdx.y * 64;
  int lane = threadIdx.x & 31;
  int rA = lane & 15;
  int kA = (lane < 16) ? 0 : 16;      // A: rows across lanes, 16 contiguous K
  int kB = (lane < 16) ? 0 : 16;      // B: cols across lanes, 16 contiguous K
  v8f acc[4];
  #pragma unroll
  for (int i = 0; i < 4; ++i) acc[i] = (v8f){0.f, 0.f, 0.f, 0.f, 0.f, 0.f, 0.f, 0.f};
  const _Float16* Abase = A + (size_t)(m0 + rA) * Kp + kA;
  const _Float16* Bb[4];
  #pragma unroll
  for (int t = 0; t < 4; ++t) Bb[t] = BT + (size_t)(n0 + t * 16 + rA) * Kp + kB;
  #pragma unroll 2
  for (int k = 0; k < Kp; k += 32) {
    const _Float16* ap = Abase + k;
    __builtin_prefetch(ap + 64, 0, 1);
    v16h af;
    #pragma unroll
    for (int e = 0; e < 16; ++e) af[e] = ap[e];
    #pragma unroll
    for (int t = 0; t < 4; ++t) {
      const _Float16* bp = Bb[t] + k;
      v16h bf;
      #pragma unroll
      for (int e = 0; e < 16; ++e) bf[e] = bp[e];
      acc[t] = __builtin_amdgcn_wmma_f32_16x16x32_f16(
          false, af, false, bf, (short)0, acc[t], false, false);
    }
  }
  int col = lane & 15;
  int rb  = (lane < 16) ? 0 : 8;
  #pragma unroll
  for (int t = 0; t < 4; ++t)
    #pragma unroll
    for (int e = 0; e < 8; ++e)
      C[(size_t)(m0 + rb + e) * ldC + n0 + t * 16 + col] = acc[t][e];
}

// ---------------------------------------------------------------- epilogues
// acc is NA x NP (stride 256); f32 activations are NA x 240; f16 operands NA x KP.
__global__ void dpse_ep_tanh(const float* __restrict__ acc, const float* __restrict__ bias,
                             float* __restrict__ xf, _Float16* __restrict__ xh) {
  int i = blockIdx.x * blockDim.x + threadIdx.x;
  if (i >= NA * KP) return;
  int row = i >> 8, c = i & 255;
  if (c < FITW) {
    float v = tanhf(acc[row * NP + c] + bias[c]);
    xf[row * FITW + c] = v;
    xh[i] = (_Float16)v;
  } else xh[i] = (_Float16)0.f;
}

__global__ void dpse_ep_res(const float* __restrict__ acc, const float* __restrict__ bias,
                            const float* __restrict__ xprev,
                            float* __restrict__ xf, _Float16* __restrict__ xh) {
  int i = blockIdx.x * blockDim.x + threadIdx.x;
  if (i >= NA * KP) return;
  int row = i >> 8, c = i & 255;
  if (c < FITW) {
    float v = tanhf(acc[row * NP + c] + bias[c]) + xprev[row * FITW + c];
    xf[row * FITW + c] = v;
    xh[i] = (_Float16)v;
  } else xh[i] = (_Float16)0.f;
}

__global__ void dpse_out_energy(const int* __restrict__ itype, const float* __restrict__ x3f,
                                const float* __restrict__ FW3, const float* __restrict__ Fb3,
                                float* __restrict__ out) {
  int row = blockIdx.x * blockDim.x + threadIdx.x;
  if (row >= NA) return;
  int t = itype[row];
  const float* x = x3f + (size_t)t * NA * FITW + (size_t)row * FITW;
  float e = Fb3[t];
  for (int c = 0; c < FITW; ++c) e += x[c] * FW3[t * FITW + c];
  out[2 + row] = e;                       // Ei
  atomicAdd(&out[row >> 10], e);          // Etot per batch
}

__global__ void dpse_bwd_init(const int* __restrict__ itype, int t,
                              const float* __restrict__ W3col,
                              const float* __restrict__ x3f, const float* __restrict__ x2f,
                              float* __restrict__ dx3, _Float16* __restrict__ dhh) {
  int i = blockIdx.x * blockDim.x + threadIdx.x;
  if (i >= NA * KP) return;
  int row = i >> 8, c = i & 255;
  if (c < FITW) {
    float g = (itype[row] == t) ? W3col[c] : 0.f;
    dx3[row * FITW + c] = g;
    float h2 = x3f[row * FITW + c] - x2f[row * FITW + c];
    dhh[i] = (_Float16)(g * (1.f - h2 * h2));
  } else dhh[i] = (_Float16)0.f;
}

__global__ void dpse_ep_bwd2(const float* __restrict__ acc, const float* __restrict__ dx3,
                             const float* __restrict__ x1f, const float* __restrict__ x2f,
                             float* __restrict__ dx2, _Float16* __restrict__ dhh) {
  int i = blockIdx.x * blockDim.x + threadIdx.x;
  if (i >= NA * KP) return;
  int row = i >> 8, c = i & 255;
  if (c < FITW) {
    float v = dx3[row * FITW + c] + acc[row * NP + c];
    dx2[row * FITW + c] = v;
    float h1 = x2f[row * FITW + c] - x1f[row * FITW + c];
    dhh[i] = (_Float16)(v * (1.f - h1 * h1));
  } else dhh[i] = (_Float16)0.f;
}

__global__ void dpse_ep_bwd1(const float* __restrict__ acc, const float* __restrict__ dx2,
                             const float* __restrict__ x1f, _Float16* __restrict__ dhh) {
  int i = blockIdx.x * blockDim.x + threadIdx.x;
  if (i >= NA * KP) return;
  int row = i >> 8, c = i & 255;
  if (c < FITW) {
    float v = dx2[row * FITW + c] + acc[row * NP + c];
    float h0 = x1f[row * FITW + c];
    dhh[i] = (_Float16)(v * (1.f - h0 * h0));
  } else dhh[i] = (_Float16)0.f;
}

// ------------------------------------------------------------ dD -> dscat
__global__ void __launch_bounds__(128)
dpse_scat_bwd(const int* __restrict__ itype, const float* __restrict__ dD,
              const float* __restrict__ scatg, float* __restrict__ dscat) {
  __shared__ float dd[DDIM];
  __shared__ float sc[400];
  int atom = blockIdx.x;
  int tid  = threadIdx.x;
  int t    = itype[atom];
  const float* dp = dD + ((size_t)t * NA + atom) * DDIM;
  const float* sp = scatg + ((size_t)t * NA + atom) * 400;
  for (int i = tid; i < DDIM; i += 128) dd[i] = dp[i];
  for (int i = tid; i < 400; i += 128) sc[i] = sp[i];
  __syncthreads();
  for (int i = tid; i < 400; i += 128) {
    int f = i / 100, g = i % 100;
    float v = 0.f;
    #pragma unroll
    for (int h = 0; h < 16; ++h) v += dd[g * 16 + h] * sc[f * 100 + h];
    if (g < 16)
      for (int gp = 0; gp < 100; ++gp) v += dd[gp * 16 + g] * sc[f * 100 + gp];
    dscat[(size_t)atom * 400 + i] = v;
  }
}

// ------------------------------------------------ embedding backward + force
__global__ void __launch_bounds__(64)
dpse_embed_bwd(const int* __restrict__ itype, const int* __restrict__ nlist,
               const float* __restrict__ Ri, const float* __restrict__ Rid,
               const float* __restrict__ dscat,
               const float* __restrict__ EW0, const float* __restrict__ Eb0,
               const float* __restrict__ EW1, const float* __restrict__ Eb1,
               const float* __restrict__ EW2, const float* __restrict__ Eb2,
               float* __restrict__ F) {
  __shared__ float sW0[25], sB0[25], sW1[25 * 50], sB1[50], sW2[50 * 100], sB2[100];
  __shared__ float sdsc[400];
  __shared__ float sx1[64 * 25], sx2[64 * 50], sdx2[64 * 50];
  int j    = blockIdx.x & 1;
  int atom = blockIdx.x >> 1;
  int m    = threadIdx.x;
  int t = itype[atom];
  int p = t * NTY + j;
  for (int i = m; i < 1250; i += 64) sW1[i] = EW1[p * 1250 + i];
  for (int i = m; i < 5000; i += 64) sW2[i] = EW2[p * 5000 + i];
  if (m < 25) { sW0[m] = EW0[p * 25 + m]; sB0[m] = Eb0[p * 25 + m]; }
  if (m < 50) sB1[m] = Eb1[p * 50 + m];
  for (int i = m; i < 100; i += 64) sB2[i] = Eb2[p * 100 + i];
  for (int i = m; i < 400; i += 64) sdsc[i] = dscat[(size_t)atom * 400 + i];
  __syncthreads();

  int nrow = atom * MN + j * MBLK + m;
  float blk[4];
  #pragma unroll
  for (int f = 0; f < 4; ++f) blk[f] = Ri[nrow * 4 + f];
  float s = blk[0];
  // recompute forward
  for (int k = 0; k < 25; ++k) sx1[m * 25 + k] = tanhf(s * sW0[k] + sB0[k]);
  for (int g = 0; g < 50; ++g) {
    float a = sB1[g];
    for (int k = 0; k < 25; ++k) a += sx1[m * 25 + k] * sW1[k * 50 + g];
    sx2[m * 50 + g] = tanhf(a) + sx1[m * 25 + (g % 25)];
  }
  for (int g = 0; g < 50; ++g) sdx2[m * 50 + g] = 0.f;
  const float invMN = 1.f / (float)MN;
  float dblk[4] = {0.f, 0.f, 0.f, 0.f};
  // layer 3 recompute + backward
  for (int g = 0; g < 100; ++g) {
    float a = sB2[g];
    for (int k = 0; k < 50; ++k) a += sx2[m * 50 + k] * sW2[k * 100 + g];
    float hB = tanhf(a);
    float G  = hB + sx2[m * 50 + (g % 50)];
    float dG = 0.f;
    #pragma unroll
    for (int f = 0; f < 4; ++f) {
      float w = sdsc[f * 100 + g] * invMN;
      dG += w * blk[f];
      dblk[f] += w * G;
    }
    float dhB = dG * (1.f - hB * hB);
    for (int k = 0; k < 50; ++k) sdx2[m * 50 + k] += dhB * sW2[k * 100 + g];
    sdx2[m * 50 + (g % 50)] += dG;   // residual concat path
  }
  // layer 2 backward
  float dx1r[25];
  #pragma unroll
  for (int k = 0; k < 25; ++k) dx1r[k] = sdx2[m * 50 + k] + sdx2[m * 50 + k + 25];
  for (int g = 0; g < 50; ++g) {
    float hA = sx2[m * 50 + g] - sx1[m * 25 + (g % 25)];
    sdx2[m * 50 + g] *= (1.f - hA * hA);
  }
  #pragma unroll
  for (int k = 0; k < 25; ++k) {
    float a = 0.f;
    for (int g = 0; g < 50; ++g) a += sdx2[m * 50 + g] * sW1[k * 50 + g];
    dx1r[k] += a;
  }
  // layer 1 backward -> ds
  float ds = 0.f;
  #pragma unroll
  for (int k = 0; k < 25; ++k) {
    float x1v = sx1[m * 25 + k];
    ds += dx1r[k] * (1.f - x1v * x1v) * sW0[k];
  }
  float dRi0 = dblk[0] + ds;
  // contract with Ri_d, scatter forces
  const float* rd = Rid + (size_t)nrow * 12;
  int bi  = atom >> 10;
  int ai  = atom & 1023;
  int idx = nlist[(size_t)atom * MN + j * MBLK + m];
  #pragma unroll
  for (int c = 0; c < 3; ++c) {
    float v = dRi0 * rd[c] + dblk[1] * rd[3 + c] + dblk[2] * rd[6 + c] + dblk[3] * rd[9 + c];
    atomicAdd(&F[(size_t)bi * NPA * 3 + ai * 3 + c], -v);          // center
    if (idx > 0)
      atomicAdd(&F[(size_t)bi * NPA * 3 + (idx - 1) * 3 + c], v);  // neighbor
  }
}

// ---------------------------------------------------------------- launcher
extern "C" void kernel_launch(void* const* d_in, const int* in_sizes, int n_in,
                              void* d_out, int out_size, void* d_ws, size_t ws_size,
                              hipStream_t stream) {
  const int*   itype = (const int*)d_in[0];
  const int*   nlist = (const int*)d_in[1];
  const float* idr   = (const float*)d_in[3];
  const float* EW0 = (const float*)d_in[5],  *Eb0 = (const float*)d_in[6];
  const float* EW1 = (const float*)d_in[7],  *Eb1 = (const float*)d_in[8];
  const float* EW2 = (const float*)d_in[9],  *Eb2 = (const float*)d_in[10];
  const float* FW0 = (const float*)d_in[11], *Fb0 = (const float*)d_in[12];
  const float* FW1 = (const float*)d_in[13], *Fb1 = (const float*)d_in[14];
  const float* FW2 = (const float*)d_in[15], *Fb2 = (const float*)d_in[16];
  const float* FW3 = (const float*)d_in[17], *Fb3 = (const float*)d_in[18];
  float* out = (float*)d_out;

  char* base = (char*)d_ws;
  size_t cur = 0;
  auto alloc = [&](size_t nbytes) -> void* {
    void* p = (void*)(base + cur);
    cur += (nbytes + 255) & ~(size_t)255;
    return p;
  };
  float* Ri    = (float*)alloc((size_t)NA * MN * 4 * 4);
  float* Rid   = (float*)alloc((size_t)NA * MN * 12 * 4);
  float* scatg = (float*)alloc((size_t)NTY * NA * 400 * 4);
  _Float16* Dh = (_Float16*)alloc((size_t)NTY * NA * DDIM * 2);
  float* x1f   = (float*)alloc((size_t)NTY * NA * FITW * 4);
  float* x2f   = (float*)alloc((size_t)NTY * NA * FITW * 4);
  float* x3f   = (float*)alloc((size_t)NTY * NA * FITW * 4);
  _Float16* x1h = (_Float16*)alloc((size_t)NA * KP * 2);
  _Float16* x2h = (_Float16*)alloc((size_t)NA * KP * 2);
  float* accb  = (float*)alloc((size_t)NA * NP * 4);
  float* dx3   = (float*)alloc((size_t)NA * FITW * 4);
  float* dx2   = (float*)alloc((size_t)NA * FITW * 4);
  _Float16* dhh = (_Float16*)alloc((size_t)NA * KP * 2);
  float* dD    = (float*)alloc((size_t)NTY * NA * DDIM * 4);
  float* dscat = (float*)alloc((size_t)NA * 400 * 4);
  _Float16* W0T = (_Float16*)alloc((size_t)NTY * NP * DDIM * 2);
  _Float16* W1T = (_Float16*)alloc((size_t)NTY * NP * KP * 2);
  _Float16* W2T = (_Float16*)alloc((size_t)NTY * NP * KP * 2);
  _Float16* W0P = (_Float16*)alloc((size_t)NTY * DDIM * KP * 2);
  _Float16* W1P = (_Float16*)alloc((size_t)NTY * NP * KP * 2);
  _Float16* W2P = (_Float16*)alloc((size_t)NTY * NP * KP * 2);

  dpse_zero<<<dim3(32), dim3(256), 0, stream>>>(out, out_size);

  for (int t = 0; t < NTY; ++t) {
    int tot;
    tot = NP * DDIM;   // W0T: 256 x 1600
    dpse_cvtT<<<dim3((tot + 255) / 256), dim3(256), 0, stream>>>(
        FW0 + (size_t)t * DDIM * FITW, W0T + (size_t)t * NP * DDIM, DDIM, FITW, DDIM, tot);
    tot = NP * KP;     // W1T/W2T: 256 x 256
    dpse_cvtT<<<dim3((tot + 255) / 256), dim3(256), 0, stream>>>(
        FW1 + (size_t)t * FITW * FITW, W1T + (size_t)t * NP * KP, FITW, FITW, KP, tot);
    dpse_cvtT<<<dim3((tot + 255) / 256), dim3(256), 0, stream>>>(
        FW2 + (size_t)t * FITW * FITW, W2T + (size_t)t * NP * KP, FITW, FITW, KP, tot);
    tot = DDIM * KP;   // W0P: 1600 x 256
    dpse_cvtP<<<dim3((tot + 255) / 256), dim3(256), 0, stream>>>(
        FW0 + (size_t)t * DDIM * FITW, W0P + (size_t)t * DDIM * KP, DDIM, FITW, KP, tot);
    tot = NP * KP;     // W1P/W2P: 256 x 256
    dpse_cvtP<<<dim3((tot + 255) / 256), dim3(256), 0, stream>>>(
        FW1 + (size_t)t * FITW * FITW, W1P + (size_t)t * NP * KP, FITW, FITW, KP, tot);
    dpse_cvtP<<<dim3((tot + 255) / 256), dim3(256), 0, stream>>>(
        FW2 + (size_t)t * FITW * FITW, W2P + (size_t)t * NP * KP, FITW, FITW, KP, tot);
  }

  dpse_calc_ri<<<dim3(NA * MN / 256), dim3(256), 0, stream>>>(idr, Ri, Rid);
  dpse_embed_fwd<<<dim3(NA * NTY), dim3(64), 0, stream>>>(
      Ri, EW0, Eb0, EW1, Eb1, EW2, Eb2, scatg, Dh);

  const dim3 gN256(NA / 16, NP / 64), gN1600(NA / 16, DDIM / 64), ep((NA * KP) / 256);
  for (int t = 0; t < NTY; ++t) {
    dpse_gemm<<<gN256, dim3(32), 0, stream>>>(
        Dh + (size_t)t * NA * DDIM, W0T + (size_t)t * NP * DDIM, accb, NP, DDIM);
    dpse_ep_tanh<<<ep, dim3(256), 0, stream>>>(accb, Fb0 + t * FITW,
        x1f + (size_t)t * NA * FITW, x1h);
    dpse_gemm<<<gN256, dim3(32), 0, stream>>>(x1h, W1T + (size_t)t * NP * KP, accb, NP, KP);
    dpse_ep_res<<<ep, dim3(256), 0, stream>>>(accb, Fb1 + t * FITW,
        x1f + (size_t)t * NA * FITW, x2f + (size_t)t * NA * FITW, x2h);
    dpse_gemm<<<gN256, dim3(32), 0, stream>>>(x2h, W2T + (size_t)t * NP * KP, accb, NP, KP);
    dpse_ep_res<<<ep, dim3(256), 0, stream>>>(accb, Fb2 + t * FITW,
        x2f + (size_t)t * NA * FITW, x3f + (size_t)t * NA * FITW, x1h /*scratch*/);
  }

  dpse_out_energy<<<dim3(NA / 256), dim3(256), 0, stream>>>(itype, x3f, FW3, Fb3, out);

  for (int t = 0; t < NTY; ++t) {
    dpse_bwd_init<<<ep, dim3(256), 0, stream>>>(itype, t, FW3 + t * FITW,
        x3f + (size_t)t * NA * FITW, x2f + (size_t)t * NA * FITW, dx3, dhh);
    dpse_gemm<<<gN256, dim3(32), 0, stream>>>(dhh, W2P + (size_t)t * NP * KP, accb, NP, KP);
    dpse_ep_bwd2<<<ep, dim3(256), 0, stream>>>(accb, dx3,
        x1f + (size_t)t * NA * FITW, x2f + (size_t)t * NA * FITW, dx2, dhh);
    dpse_gemm<<<gN256, dim3(32), 0, stream>>>(dhh, W1P + (size_t)t * NP * KP, accb, NP, KP);
    dpse_ep_bwd1<<<ep, dim3(256), 0, stream>>>(accb, dx2, x1f + (size_t)t * NA * FITW, dhh);
    dpse_gemm<<<gN1600, dim3(32), 0, stream>>>(dhh, W0P + (size_t)t * DDIM * KP,
        dD + (size_t)t * NA * DDIM, DDIM, KP);
  }

  dpse_scat_bwd<<<dim3(NA), dim3(128), 0, stream>>>(itype, dD, scatg, dscat);
  dpse_embed_bwd<<<dim3(NA * NTY), dim3(64), 0, stream>>>(
      itype, nlist, Ri, Rid, dscat, EW0, Eb0, EW1, Eb1, EW2, Eb2, out + 2 + NA);
}
